// LSTM_AE_88923002896480
// MI455X (gfx1250) — matrix operand order repaired and checked by reading, not compiled, not run on hardware
//
#include <hip/hip_runtime.h>
#include <hip/hip_bf16.h>

// ---------------------------------------------------------------------------
// LSTM autoencoder: B=1024, T=100, H=512, IN=4, gates G=4H=2048.
// Recurrent GEMMs run as bf16 WMMA (v_wmma_f32_16x16x32_bf16, f32 accum).
// ---------------------------------------------------------------------------

#define BATCH 1024
#define SEQT  100
#define HID   512
#define GATES 2048   // 4*HID
#define INDIM 4

typedef __attribute__((ext_vector_type(16))) __bf16 v16bf;
typedef __attribute__((ext_vector_type(8)))  float  v8f;

union Frag16 { uint4 u[2]; v16bf v; };

// ---------------------------------------------------------------------------
// gates[B, 2048] = A0[B,512] @ W0[2048,512]^T  (+ A1 @ W1^T if A1 != null)
// bf16 inputs, f32 accumulation via WMMA. Block tile 128x128, 8 waves,
// wave tile 32x64 (2x4 WMMA 16x16x32 accumulators).
// ---------------------------------------------------------------------------
__global__ __launch_bounds__(256)
void gemm_gates_kernel(const __bf16* __restrict__ A0, const __bf16* __restrict__ W0,
                       const __bf16* __restrict__ A1, const __bf16* __restrict__ W1,
                       float* __restrict__ gates)
{
    __shared__ __bf16 lA[128][40];   // 128 rows x 32 halves (+8 pad), 16B-aligned rows
    __shared__ __bf16 lB[128][40];

    const int K = HID;
    const int bm = blockIdx.x * 128;
    const int bn = blockIdx.y * 128;
    const int tid  = threadIdx.x;
    const int wave = tid >> 5;
    const int lane = tid & 31;
    const int l15  = lane & 15;
    const bool hi  = lane >= 16;
    const int wm = (wave >> 1) * 32;   // wave M offset within block tile
    const int wn = (wave & 1) * 64;    // wave N offset within block tile

    v8f acc[2][4] = {};                // 2 M-tiles x 4 N-tiles, f32 accum

    const int ldRow = tid >> 1;        // 128 rows, 2 threads/row
    const int ldCol = (tid & 1) * 16;  // halves 0..15 or 16..31

    for (int p = 0; p < 2; ++p) {
        const __bf16* A = p ? A1 : A0;
        const __bf16* W = p ? W1 : W0;
        if (A == nullptr) break;

        for (int k0 = 0; k0 < K; k0 += 32) {
            // ---- global -> registers (prefetch before LDS barrier) ----
            const uint4* ga = (const uint4*)(A + (size_t)(bm + ldRow) * K + k0 + ldCol);
            const uint4* gb = (const uint4*)(W + (size_t)(bn + ldRow) * K + k0 + ldCol);
            uint4 a0 = ga[0], a1 = ga[1];
            uint4 b0 = gb[0], b1 = gb[1];

            __syncthreads();           // previous tile fully consumed
            *(uint4*)&lA[ldRow][ldCol]     = a0;
            *(uint4*)&lA[ldRow][ldCol + 8] = a1;
            *(uint4*)&lB[ldRow][ldCol]     = b0;
            *(uint4*)&lB[ldRow][ldCol + 8] = b1;
            __syncthreads();

            // ---- A fragments: lane<16 holds {K0-7,K16-23}, lane>=16 {K8-15,K24-31}
            Frag16 af[2];
            #pragma unroll
            for (int i = 0; i < 2; ++i) {
                const int row = wm + i * 16 + l15;
                const int kb  = hi ? 8 : 0;
                af[i].u[0] = *(const uint4*)&lA[row][kb];
                af[i].u[1] = *(const uint4*)&lA[row][kb + 16];
            }

            // ---- B fragments: lane n holds W[n][k], lane<16 {K0-7,K16-23}, lane>=16 {K8-15,K24-31}
            #pragma unroll
            for (int j = 0; j < 4; ++j) {
                const int n  = wn + j * 16 + l15;
                const int kb = hi ? 8 : 0;
                Frag16 bfg;
                bfg.u[0] = *(const uint4*)&lB[n][kb];
                bfg.u[1] = *(const uint4*)&lB[n][kb + 16];
                #pragma unroll
                for (int i = 0; i < 2; ++i) {
                    acc[i][j] = __builtin_amdgcn_wmma_f32_16x16x32_bf16(
                        /*neg_a=*/false, af[i].v, /*neg_b=*/false, bfg.v,
                        /*c_mod=*/(short)0, acc[i][j],
                        /*reuse_a=*/false, /*reuse_b=*/false);
                }
            }
        }
    }

    // ---- store: C/D layout: VGPR r -> M = r + (lane>=16 ? 8 : 0), N = lane&15
    #pragma unroll
    for (int i = 0; i < 2; ++i) {
        const int mbase = bm + wm + i * 16 + (hi ? 8 : 0);
        #pragma unroll
        for (int j = 0; j < 4; ++j) {
            const int n = bn + wn + j * 16 + l15;
            #pragma unroll
            for (int r = 0; r < 8; ++r)
                gates[(size_t)(mbase + r) * GATES + n] = acc[i][j][r];
        }
    }
}

// ---------------------------------------------------------------------------
// Pointwise LSTM cell update. Fuses the K=4 x-projection (Wih [2048,4]) + bias
// into the gate pre-activations, then i/f/g/o nonlinearity + c/h update.
// One thread per (b, j). Writes h as f32 and bf16.
// ---------------------------------------------------------------------------
__device__ __forceinline__ float sigm(float x) { return 1.0f / (1.0f + __expf(-x)); }

__global__ __launch_bounds__(256)
void lstm_pointwise_kernel(const float* __restrict__ gates,
                           const float* __restrict__ xsrc, int xstride,
                           const float* __restrict__ Wih,   // [2048,4] or null
                           const float* __restrict__ bias,  // [2048]
                           float* __restrict__ c, float* __restrict__ h,
                           __bf16* __restrict__ hbf)
{
    const int idx = blockIdx.x * blockDim.x + threadIdx.x;  // < 1024*512
    const int b = idx >> 9;
    const int j = idx & (HID - 1);

    float x0 = 0.f, x1 = 0.f, x2 = 0.f, x3 = 0.f;
    if (Wih) {
        const float* xp = xsrc + (size_t)b * xstride;
        x0 = xp[0]; x1 = xp[1]; x2 = xp[2]; x3 = xp[3];
    }

    float g[4];
    #pragma unroll
    for (int q = 0; q < 4; ++q) {
        const int col = q * HID + j;
        float v = gates[(size_t)b * GATES + col] + bias[col];
        if (Wih) {
            const float* wr = Wih + (size_t)col * INDIM;
            v += x0 * wr[0] + x1 * wr[1] + x2 * wr[2] + x3 * wr[3];
        }
        g[q] = v;
    }

    const float ig = sigm(g[0]);
    const float fg = sigm(g[1]);
    const float gg = tanhf(g[2]);
    const float og = sigm(g[3]);
    const float cn = fg * c[idx] + ig * gg;
    const float hn = og * tanhf(cn);
    c[idx]   = cn;
    h[idx]   = hn;
    hbf[idx] = (__bf16)hn;
}

// ---------------------------------------------------------------------------
// Decoder head: pred = h1 @ Wp^T + bp  ->  logits to out[b,t,:], softmax to
// the feedback buffer. One thread per batch row (K=512, N=4).
// ---------------------------------------------------------------------------
__global__ __launch_bounds__(256)
void pred_softmax_kernel(const float* __restrict__ h1, const float* __restrict__ Wp,
                         const float* __restrict__ bp, float* __restrict__ out,
                         float* __restrict__ xfb, int t)
{
    const int b = blockIdx.x * blockDim.x + threadIdx.x;   // < 1024
    const float* hr = h1 + (size_t)b * HID;

    float lg[4];
    #pragma unroll
    for (int q = 0; q < 4; ++q) {
        const float* wr = Wp + (size_t)q * HID;
        float s = bp[q];
        for (int k = 0; k < HID; ++k) s += hr[k] * wr[k];
        lg[q] = s;
    }

    const float mx = fmaxf(fmaxf(lg[0], lg[1]), fmaxf(lg[2], lg[3]));
    float e[4], sum = 0.f;
    #pragma unroll
    for (int q = 0; q < 4; ++q) { e[q] = __expf(lg[q] - mx); sum += e[q]; }
    const float inv = 1.0f / sum;

    #pragma unroll
    for (int q = 0; q < 4; ++q) {
        xfb[(size_t)b * 4 + q] = e[q] * inv;
        out[((size_t)b * SEQT + t) * 4 + q] = lg[q];
    }
}

// ---------------------------------------------------------------------------
// f32 -> bf16 weight conversion (re-run each launch; deterministic).
// ---------------------------------------------------------------------------
__global__ __launch_bounds__(256)
void cvt_bf16_kernel(const float* __restrict__ src, __bf16* __restrict__ dst, int n)
{
    const int i = blockIdx.x * blockDim.x + threadIdx.x;
    if (i < n) dst[i] = (__bf16)src[i];
}

// ---------------------------------------------------------------------------
extern "C" void kernel_launch(void* const* d_in, const int* in_sizes, int n_in,
                              void* d_out, int out_size, void* d_ws, size_t ws_size,
                              hipStream_t stream)
{
    (void)in_sizes; (void)n_in; (void)out_size; (void)ws_size;

    const float* X        = (const float*)d_in[0];
    const float* enc_Wih0 = (const float*)d_in[2];
    const float* enc_Whh0 = (const float*)d_in[3];
    const float* enc_b0   = (const float*)d_in[4];
    const float* enc_Wih1 = (const float*)d_in[5];
    const float* enc_Whh1 = (const float*)d_in[6];
    const float* enc_b1   = (const float*)d_in[7];
    const float* dec_Wih0 = (const float*)d_in[8];
    const float* dec_Whh0 = (const float*)d_in[9];
    const float* dec_b0   = (const float*)d_in[10];
    const float* dec_Wih1 = (const float*)d_in[11];
    const float* dec_Whh1 = (const float*)d_in[12];
    const float* dec_b1   = (const float*)d_in[13];
    const float* Wp       = (const float*)d_in[14];
    const float* bp       = (const float*)d_in[15];
    float* out = (float*)d_out;

    // ---- workspace carve-out (256B aligned) ----
    char* w = (char*)d_ws;
    auto carve = [&](size_t bytes) -> void* {
        void* p = (void*)w;
        w += (bytes + 255) & ~(size_t)255;
        return p;
    };
    const size_t WSZ = (size_t)GATES * HID;           // 1M elems per big weight
    __bf16* bWhh0e = (__bf16*)carve(WSZ * 2);
    __bf16* bWih1e = (__bf16*)carve(WSZ * 2);
    __bf16* bWhh1e = (__bf16*)carve(WSZ * 2);
    __bf16* bWhh0d = (__bf16*)carve(WSZ * 2);
    __bf16* bWih1d = (__bf16*)carve(WSZ * 2);
    __bf16* bWhh1d = (__bf16*)carve(WSZ * 2);
    float*  gates  = (float*)carve((size_t)BATCH * GATES * 4);
    float*  h0     = (float*)carve((size_t)BATCH * HID * 4);
    float*  c0     = (float*)carve((size_t)BATCH * HID * 4);
    float*  h1     = (float*)carve((size_t)BATCH * HID * 4);
    float*  c1     = (float*)carve((size_t)BATCH * HID * 4);
    __bf16* h0bf   = (__bf16*)carve((size_t)BATCH * HID * 2);
    __bf16* h1bf   = (__bf16*)carve((size_t)BATCH * HID * 2);
    float*  xfb    = (float*)carve((size_t)BATCH * 4 * 4);

    const dim3 cvtGrid((WSZ + 255) / 256), blk(256);
    cvt_bf16_kernel<<<cvtGrid, blk, 0, stream>>>(enc_Whh0, bWhh0e, (int)WSZ);
    cvt_bf16_kernel<<<cvtGrid, blk, 0, stream>>>(enc_Wih1, bWih1e, (int)WSZ);
    cvt_bf16_kernel<<<cvtGrid, blk, 0, stream>>>(enc_Whh1, bWhh1e, (int)WSZ);
    cvt_bf16_kernel<<<cvtGrid, blk, 0, stream>>>(dec_Whh0, bWhh0d, (int)WSZ);
    cvt_bf16_kernel<<<cvtGrid, blk, 0, stream>>>(dec_Wih1, bWih1d, (int)WSZ);
    cvt_bf16_kernel<<<cvtGrid, blk, 0, stream>>>(dec_Whh1, bWhh1d, (int)WSZ);

    // zero initial states (graph-capture-safe memset nodes)
    hipMemsetAsync(h0,   0, (size_t)BATCH * HID * 4, stream);
    hipMemsetAsync(c0,   0, (size_t)BATCH * HID * 4, stream);
    hipMemsetAsync(h1,   0, (size_t)BATCH * HID * 4, stream);
    hipMemsetAsync(c1,   0, (size_t)BATCH * HID * 4, stream);
    hipMemsetAsync(h0bf, 0, (size_t)BATCH * HID * 2, stream);
    hipMemsetAsync(h1bf, 0, (size_t)BATCH * HID * 2, stream);

    const dim3 gemmGrid(BATCH / 128, GATES / 128);        // 8 x 16
    const dim3 pwGrid((size_t)BATCH * HID / 256);         // 2048
    const dim3 predGrid(BATCH / 256);                     // 4

    // ---------------- encoder: 2-layer LSTM over T steps ----------------
    for (int t = 0; t < SEQT; ++t) {
        // layer 0: gates = h0 @ Whh0^T ; x-contrib (K=4) fused in pointwise
        gemm_gates_kernel<<<gemmGrid, blk, 0, stream>>>(h0bf, bWhh0e, nullptr, nullptr, gates);
        lstm_pointwise_kernel<<<pwGrid, blk, 0, stream>>>(
            gates, X + (size_t)t * INDIM, SEQT * INDIM, enc_Wih0, enc_b0, c0, h0, h0bf);
        // layer 1: gates = h0 @ Wih1^T + h1 @ Whh1^T
        gemm_gates_kernel<<<gemmGrid, blk, 0, stream>>>(h0bf, bWih1e, h1bf, bWhh1e, gates);
        lstm_pointwise_kernel<<<pwGrid, blk, 0, stream>>>(
            gates, nullptr, 0, nullptr, enc_b1, c1, h1, h1bf);
    }

    // ---------------- decoder: autoregressive with softmax feedback ----------------
    hipMemsetAsync(xfb, 0, (size_t)BATCH * 4 * 4, stream);   // x_init = zeros
    for (int t = 0; t < SEQT; ++t) {
        gemm_gates_kernel<<<gemmGrid, blk, 0, stream>>>(h0bf, bWhh0d, nullptr, nullptr, gates);
        lstm_pointwise_kernel<<<pwGrid, blk, 0, stream>>>(
            gates, xfb, INDIM, dec_Wih0, dec_b0, c0, h0, h0bf);
        gemm_gates_kernel<<<gemmGrid, blk, 0, stream>>>(h0bf, bWih1d, h1bf, bWhh1d, gates);
        lstm_pointwise_kernel<<<pwGrid, blk, 0, stream>>>(
            gates, nullptr, 0, nullptr, dec_b1, c1, h1, h1bf);
        pred_softmax_kernel<<<predGrid, blk, 0, stream>>>(h1, Wp, bp, out, xfb, t);
    }
}
